// MeshDeformationBlock_44976897523721
// MI455X (gfx1250) — matrix operand order repaired and checked by reading, not compiled, not run on hardware
//
#include <hip/hip_runtime.h>
#include <hip/hip_bf16.h>

typedef __attribute__((ext_vector_type(16))) __bf16 bf16x16;
typedef __attribute__((ext_vector_type(8)))  float  f32x8;

#define N_NODES 10000
#define N_EDGES 80000
#define E2      90000          // edges + self loops
#define MPAD    10112          // 10000 rounded up to multiple of 128
#define KPMAX   2368           // 2307 rounded up to multiple of 64
#define NTPMAX  6144           // max 2*H*cout, multiple of 128
#define HCMAX   3072

static __device__ __forceinline__ unsigned fmap(float f) {
    unsigned u = __float_as_uint(f);
    return (u & 0x80000000u) ? ~u : (u | 0x80000000u);
}
static __device__ __forceinline__ float funmap(unsigned m) {
    unsigned u = (m & 0x80000000u) ? (m & 0x7fffffffu) : ~m;
    return __uint_as_float(u);
}
static __device__ __forceinline__ void get_edge(const int* __restrict__ ei, int e,
                                                int& s, int& d) {
    if (e < N_EDGES) { s = ei[e]; d = ei[N_EDGES + e]; }
    else             { s = d = e - N_EDGES; }
}
static __device__ __forceinline__ bf16x16 load_frag(const __bf16* __restrict__ p) {
    union { bf16x16 v; uint4 u[2]; } f;
    f.u[0] = *(const uint4*)(p);
    f.u[1] = *(const uint4*)(p + 16);
    return f.v;
}

// ---------- activation f32 -> bf16, padded [MPAD x Kp] ----------
__global__ void cvt_act_kernel(const float* __restrict__ x, __bf16* __restrict__ act,
                               int cin, int Kp) {
    int t = blockIdx.x * blockDim.x + threadIdx.x;
    int total = MPAD * Kp;
    if (t >= total) return;
    int m = t / Kp, k = t - m * Kp;
    float v = (m < N_NODES && k < cin) ? x[(size_t)m * cin + k] : 0.0f;
    act[t] = (__bf16)v;
}

// ---------- weights: WT[n][k] = (n<ncols ? Wl[k][n] : Wr[k][n-ncols]), padded ----------
__global__ void cvt_w_kernel(const float* __restrict__ Wl, const float* __restrict__ Wr,
                             __bf16* __restrict__ wt, int cin, int ncols, int Kp, int NtotPad) {
    int t = blockIdx.x * blockDim.x + threadIdx.x;
    int total = NtotPad * Kp;
    if (t >= total) return;
    int n = t / Kp, k = t - n * Kp;
    float v = 0.0f;
    if (k < cin) {
        if (n < ncols)            v = Wl[(size_t)k * ncols + n];
        else if (n < 2 * ncols)   v = Wr[(size_t)k * ncols + (n - ncols)];
    }
    wt[t] = (__bf16)v;
}

// ---------- WMMA GEMM: C[MPAD x NtotPad] = A[MPAD x Kp] * WT^T ----------
// block = 256 threads = 8 waves tiling 128(M) x 128(N); each wave: 32(M) x 64(N).
// Double-buffered k-loop: loads for step i+2 issued right after WMMAs of step i.
__global__ __launch_bounds__(256)
void gemm_wmma_kernel(const __bf16* __restrict__ A, const __bf16* __restrict__ B,
                      float* __restrict__ C, int Kp, int ldx) {
    const int lane = threadIdx.x & 31;
    const int wave = threadIdx.x >> 5;
    const int m0 = blockIdx.x * 128 + (wave >> 1) * 32;
    const int n0 = blockIdx.y * 128 + (wave & 1) * 64;
    const int l  = lane & 15;
    const bool hi = lane >= 16;

    f32x8 acc[8] = {};

    // A fragment rows: lane<16 -> K {0..7,16..23}; lane>=16 -> K {8..15,24..31}
    const __bf16* ap0 = A + (size_t)(m0 + l) * Kp + (hi ? 8 : 0);
    const __bf16* ap1 = ap0 + (size_t)16 * Kp;
    // B fragment rows: lane<16 -> K {0..7,16..23}; lane>=16 -> K {8..15,24..31}
    const __bf16* bp0 = B + (size_t)(n0 + l) * Kp + (hi ? 8 : 0);
    const __bf16* bp1 = bp0 + (size_t)16 * Kp;
    const __bf16* bp2 = bp0 + (size_t)32 * Kp;
    const __bf16* bp3 = bp0 + (size_t)48 * Kp;

    bf16x16 a0[2], b0[4], a1[2], b1[4];

#define LOADG(AA, BB, kk)                                                      \
    do {                                                                       \
        AA[0] = load_frag(ap0 + (kk));  AA[1] = load_frag(ap1 + (kk));         \
        BB[0] = load_frag(bp0 + (kk));  BB[1] = load_frag(bp1 + (kk));         \
        BB[2] = load_frag(bp2 + (kk));  BB[3] = load_frag(bp3 + (kk));         \
    } while (0)

#define MMA8(AA, BB)                                                           \
    do {                                                                       \
        _Pragma("unroll")                                                      \
        for (int mt = 0; mt < 2; ++mt)                                         \
            _Pragma("unroll")                                                  \
            for (int j = 0; j < 4; ++j)                                        \
                acc[mt * 4 + j] = __builtin_amdgcn_wmma_f32_16x16x32_bf16(     \
                    false, AA[mt], false, BB[j], (short)0, acc[mt * 4 + j],    \
                    false, false);                                             \
    } while (0)

    // Kp is a multiple of 64 -> even number of 32-wide k-steps (>= 2)
    LOADG(a0, b0, 0);
    LOADG(a1, b1, 32);
    for (int k = 64; k < Kp; k += 64) {
        MMA8(a0, b0);
        LOADG(a0, b0, k);
        MMA8(a1, b1);
        LOADG(a1, b1, k + 32);
    }
    MMA8(a0, b0);
    MMA8(a1, b1);

#undef LOADG
#undef MMA8

    // C/D layout: lane<16 -> N=lane, M rows 0..7; lane>=16 -> N=lane-16, M rows 8..15
    const int mbase = m0 + (hi ? 8 : 0);
#pragma unroll
    for (int mt = 0; mt < 2; ++mt)
#pragma unroll
        for (int j = 0; j < 4; ++j)
#pragma unroll
            for (int r = 0; r < 8; ++r)
                C[(size_t)(mbase + mt * 16 + r) * ldx + (n0 + 16 * j + l)] =
                    acc[mt * 4 + j][r];
}

// ---------- edge logits: wave per (edge, head), reduce over channels ----------
__global__ __launch_bounds__(256)
void edge_logits_kernel(const float* __restrict__ xlxr, const int* __restrict__ ei,
                        const float* __restrict__ att, float* __restrict__ logits,
                        unsigned* __restrict__ maxbuf, int H, int C, int ldx, int ncols) {
    int wid  = (blockIdx.x * blockDim.x + threadIdx.x) >> 5;
    int lane = threadIdx.x & 31;
    if (wid >= E2 * H) return;
    int e = wid / H, h = wid - e * H;
    int s, d; get_edge(ei, e, s, d);

    const float* pl = xlxr + (size_t)s * ldx + h * C;            // xl[src]
    const float* pr = xlxr + (size_t)d * ldx + ncols + h * C;    // xr[dst]
    const float* pa = att + h * C;
    float sum = 0.0f;
    for (int c = lane; c < C; c += 32) {
        float v = pl[c] + pr[c];
        v = v > 0.0f ? v : 0.2f * v;          // leaky_relu
        sum += v * pa[c];
    }
#pragma unroll
    for (int off = 16; off > 0; off >>= 1) sum += __shfl_xor(sum, off, 32);
    if (lane == 0) {
        logits[wid] = sum;
        atomicMax(&maxbuf[d * H + h], fmap(sum));
    }
}

// ---------- a = exp(logit - max[dst]); denom[dst] += a ----------
__global__ void edge_exp_kernel(float* __restrict__ alpha, const unsigned* __restrict__ maxbuf,
                                float* __restrict__ denom, const int* __restrict__ ei, int H) {
    int t = blockIdx.x * blockDim.x + threadIdx.x;
    if (t >= E2 * H) return;
    int e = t / H, h = t - e * H;
    int s, d; get_edge(ei, e, s, d); (void)s;
    float a = __expf(alpha[t] - funmap(maxbuf[d * H + h]));
    alpha[t] = a;
    atomicAdd(&denom[d * H + h], a);
}

__global__ void edge_norm_kernel(float* __restrict__ alpha, const float* __restrict__ denom,
                                 const int* __restrict__ ei, int H) {
    int t = blockIdx.x * blockDim.x + threadIdx.x;
    if (t >= E2 * H) return;
    int e = t / H, h = t - e * H;
    int s, d; get_edge(ei, e, s, d); (void)s;
    alpha[t] = alpha[t] / denom[d * H + h];
}

// ---------- scatter: out[dst][h][c] += alpha[e][h] * xl[src][h][c] ----------
__global__ void scatter_kernel(const float* __restrict__ alpha, const float* __restrict__ xlxr,
                               float* __restrict__ outacc, const int* __restrict__ ei,
                               int H, int C, int ldx) {
    int t = blockIdx.x * blockDim.x + threadIdx.x;
    int HC = H * C;
    if (t >= E2 * HC) return;
    int e = t / HC, r = t - e * HC;
    int h = r / C;
    int s, d; get_edge(ei, e, s, d);
    float v = alpha[e * H + h] * xlxr[(size_t)s * ldx + r];
    atomicAdd(&outacc[(size_t)d * HC + r], v);
}

// ---------- finalize: mean over heads + bias (+relu), write bf16 act (+ f32 out) ----------
__global__ void finalize_kernel(const float* __restrict__ outacc, const float* __restrict__ bias,
                                __bf16* __restrict__ act, float* __restrict__ outp,
                                int C, int H, int KpNext, float invH) {
    int t = blockIdx.x * blockDim.x + threadIdx.x;
    if (t >= MPAD * KpNext) return;
    int m = t / KpNext, c = t - m * KpNext;
    float v = 0.0f;
    if (m < N_NODES && c < C) {
        int HC = H * C;
        float s = 0.0f;
        for (int h = 0; h < H; ++h) s += outacc[(size_t)m * HC + h * C + c];
        v = s * invH + bias[c];
        v = fmaxf(v, 0.0f);                       // ReLU (all non-final layers)
        if (outp) outp[(size_t)m * C + c] = v;    // layer-7 output goes to d_out too
    }
    act[t] = (__bf16)v;
}

__global__ void finalize_last_kernel(const float* __restrict__ outacc,
                                     const float* __restrict__ bias,
                                     float* __restrict__ coords) {
    int t = blockIdx.x * blockDim.x + threadIdx.x;
    if (t >= N_NODES * 3) return;
    int c = t % 3;
    coords[t] = outacc[t] + bias[c];              // H=1, concat, no activation
}

static inline int rup(int v, int a) { return (v + a - 1) / a * a; }

extern "C" void kernel_launch(void* const* d_in, const int* in_sizes, int n_in,
                              void* d_out, int out_size, void* d_ws, size_t ws_size,
                              hipStream_t stream) {
    (void)in_sizes; (void)n_in; (void)out_size; (void)ws_size;

    static const int CIN [8] = {2307, 1024, 1024, 512, 512, 256, 256, 128};
    static const int COUT[8] = {1024, 1024,  512, 512, 256, 256, 128,   3};
    static const int HH  [8] = {3, 3, 3, 3, 3, 3, 3, 1};

    const float* x  = (const float*)d_in[0];
    const int*   ei = (const int*)d_in[1];

    char* ws = (char*)d_ws;
    size_t o_act  = 0;
    size_t o_wt   = o_act  + (size_t)MPAD   * KPMAX * 2;
    size_t o_xlxr = o_wt   + (size_t)NTPMAX * KPMAX * 2;
    size_t o_out  = o_xlxr + (size_t)MPAD   * NTPMAX * 4;
    size_t o_alp  = o_out  + (size_t)N_NODES * HCMAX * 4;
    size_t o_max  = o_alp  + (((size_t)E2 * 3 * 4 + 255) & ~(size_t)255);
    size_t o_den  = o_max  + (((size_t)N_NODES * 3 * 4 + 255) & ~(size_t)255);

    __bf16*   ACT    = (__bf16*)(ws + o_act);
    __bf16*   WT     = (__bf16*)(ws + o_wt);
    float*    XLXR   = (float*)(ws + o_xlxr);
    float*    OUTACC = (float*)(ws + o_out);
    float*    ALPHA  = (float*)(ws + o_alp);
    unsigned* MAXBUF = (unsigned*)(ws + o_max);
    float*    DENOM  = (float*)(ws + o_den);
    float*    dout   = (float*)d_out;

    const int B = 256;

    // Layer-0 input: f32 -> padded bf16
    {
        int Kp = rup(CIN[0], 64);
        int total = MPAD * Kp;
        cvt_act_kernel<<<(total + B - 1) / B, B, 0, stream>>>(x, ACT, CIN[0], Kp);
    }

    for (int L = 0; L < 8; ++L) {
        const int cin = CIN[L], cout = COUT[L], H = HH[L];
        const int ncols = H * cout;
        const int Kp = rup(cin, 64);
        const int NtotPad = rup(2 * ncols, 128);
        const int HC = H * cout;
        const float* Wl   = (const float*)d_in[2 + L * 4 + 0];
        const float* Wr   = (const float*)d_in[2 + L * 4 + 1];
        const float* att  = (const float*)d_in[2 + L * 4 + 2];
        const float* bias = (const float*)d_in[2 + L * 4 + 3];

        // weights -> transposed padded bf16 [NtotPad x Kp]
        {
            int total = NtotPad * Kp;
            cvt_w_kernel<<<(total + B - 1) / B, B, 0, stream>>>(Wl, Wr, WT, cin, ncols, Kp, NtotPad);
        }

        // fused GEMM: [xl | xr] = act @ [Wl | Wr]
        {
            dim3 g(MPAD / 128, NtotPad / 128);
            gemm_wmma_kernel<<<g, B, 0, stream>>>(ACT, WT, XLXR, Kp, NtotPad);
        }

        hipMemsetAsync(OUTACC, 0, (size_t)N_NODES * HC * 4, stream);
        hipMemsetAsync(MAXBUF, 0, (size_t)N_NODES * H * 4, stream);
        hipMemsetAsync(DENOM,  0, (size_t)N_NODES * H * 4, stream);

        const int tasks = E2 * H;
        edge_logits_kernel<<<(tasks + 7) / 8, B, 0, stream>>>(XLXR, ei, att, ALPHA, MAXBUF,
                                                              H, cout, NtotPad, ncols);
        edge_exp_kernel<<<(tasks + B - 1) / B, B, 0, stream>>>(ALPHA, MAXBUF, DENOM, ei, H);
        edge_norm_kernel<<<(tasks + B - 1) / B, B, 0, stream>>>(ALPHA, DENOM, ei, H);

        {
            int total = E2 * HC;   // max 2.76e8, fits int
            scatter_kernel<<<(total + B - 1) / B, B, 0, stream>>>(ALPHA, XLXR, OUTACC, ei,
                                                                  H, cout, NtotPad);
        }

        if (L < 7) {
            const int KpNext = cout;                 // next cin == cout, already mult of 64
            float* outp = (L == 6) ? dout : nullptr; // layer-7 ReLU output is first d_out chunk
            int total = MPAD * KpNext;
            finalize_kernel<<<(total + B - 1) / B, B, 0, stream>>>(OUTACC, bias, ACT, outp,
                                                                   cout, H, KpNext, 1.0f / H);
        } else {
            int total = N_NODES * 3;
            finalize_last_kernel<<<(total + B - 1) / B, B, 0, stream>>>(OUTACC, bias,
                                                                        dout + (size_t)N_NODES * 128);
        }
    }
}
